// TransformerLanguageModel_22797686408041
// MI455X (gfx1250) — matrix operand order repaired and checked by reading, not compiled, not run on hardware
//
#include <hip/hip_runtime.h>
#include <hip/hip_bf16.h>
#include <math.h>

typedef _Float16 v2h  __attribute__((ext_vector_type(2)));
typedef _Float16 v8h  __attribute__((ext_vector_type(8)));
typedef _Float16 v16h __attribute__((ext_vector_type(16)));
typedef float    v8f  __attribute__((ext_vector_type(8)));

#define Dm   512
#define Tm   128
#define Hm   8
#define DHm  64
#define Lm   12
#define Vm   200018
#define Bbat 8
#define Mrows 1024   // B*T

// ---------------- WMMA GEMM ----------------
// C[z] = act( A[z] * B[z] + bias ) (+ resid)  with f32->f16 staging in LDS.
// z offset = (z/zdiv)*stride1 + (z%zdiv)*stride2 for A, B, C (resid uses C offsets).
// Grid: blockIdx.x = M tile (fast-dispatch axis -> B slabs reused in L2 across M tiles),
//       blockIdx.y = N tile, blockIdx.z = batch.
// Waves: 4x2 grid; each wave computes 32 rows x (BN_/2) cols:
//   2 A fragments, BN_/32 B fragments, 2*(BN_/32) wmmas per K-step.
// Register double-buffering (next K-tile global loads issued before the wmma burst),
// paired K elements so staging uses b64 global loads + packed b32 LDS stores.
#define BM 128
#define BK 32
#define ASTR 40
#define BSTR 40

template<int BN_>
__global__ void __launch_bounds__(256)
gemm_wmma(const float* __restrict__ A, int lda, long long sA1, long long sA2,
          const float* __restrict__ Bmat, int ldb, long long sB1, long long sB2, int transB,
          float* __restrict__ C, int ldc, long long sC1, long long sC2,
          const float* __restrict__ bias, const float* __restrict__ resid,
          int M, int N, int K, int zdiv, int relu)
{
  constexpr int NAP = BM * BK / 512;   // A float2-pairs per thread per K-tile (8)
  constexpr int NBP = BN_ * BK / 512;  // B pairs per thread per K-tile (4 or 8)
  constexpr int NJ  = BN_ / 32;        // 16-col fragments per wave

  __shared__ __align__(16) _Float16 Ash[BM * ASTR];
  __shared__ __align__(16) _Float16 Bsh[BN_ * BSTR];

  int z  = blockIdx.z;
  int zq = z / zdiv, zr = z % zdiv;
  const float* Ab = A    + (long long)zq * sA1 + (long long)zr * sA2;
  const float* Bb = Bmat + (long long)zq * sB1 + (long long)zr * sB2;
  float*       Cb = C    + (long long)zq * sC1 + (long long)zr * sC2;
  const float* Rb = resid ? (resid + (long long)zq * sC1 + (long long)zr * sC2) : nullptr;

  int tid  = threadIdx.x;
  int wave = tid >> 5, lane = tid & 31;
  int mm = lane & 15, hi = lane >> 4;
  int wr = wave >> 1, wc = wave & 1;           // 4x2 wave grid
  int tileM = blockIdx.x * BM, tileN = blockIdx.y * BN_;

  v8f acc[2][NJ];
#pragma unroll
  for (int i = 0; i < 2; ++i)
#pragma unroll
    for (int j = 0; j < NJ; ++j) {
      v8f zv = {0.f, 0.f, 0.f, 0.f, 0.f, 0.f, 0.f, 0.f};
      acc[i][j] = zv;
    }

  float2 aP[NAP];
  float2 bP[NBP];

  // A pair mapping: u = tid + p*256 over 2048 pairs; 16 pairs per 32-wide K row.
  auto loadA = [&](int k0) {
#pragma unroll
    for (int p = 0; p < NAP; ++p) {
      int u = tid + p * 256;
      int r = u >> 4, c = (u & 15) * 2;        // consecutive lanes -> consecutive float2
      aP[p] = *(const float2*)&Ab[(size_t)(tileM + r) * lda + (k0 + c)];
    }
  };
  auto loadB = [&](int k0) {
#pragma unroll
    for (int p = 0; p < NBP; ++p) {
      int u = tid + p * 256;
      if (transB) {
        int n = u >> 4, c = (u & 15) * 2;      // float2 along K, coalesced per row of B^T
        int gn = tileN + n;
        if (gn < N) bP[p] = *(const float2*)&Bb[(size_t)gn * ldb + (k0 + c)];
        else        bP[p] = make_float2(0.f, 0.f);
      } else {
        int n = u % BN_, kp = u / BN_;         // n fastest across lanes -> coalesced
        int k = kp * 2;
        int gn = tileN + n;
        if (gn < N) {
          bP[p].x = Bb[(size_t)(k0 + k) * ldb + gn];
          bP[p].y = Bb[(size_t)(k0 + k + 1) * ldb + gn];
        } else bP[p] = make_float2(0.f, 0.f);
      }
    }
  };

  loadA(0);
  loadB(0);

  for (int k0 = 0; k0 < K; k0 += BK) {
    __syncthreads();
    // Commit staged registers to LDS as packed f16 pairs (single b32 DS store each)
#pragma unroll
    for (int p = 0; p < NAP; ++p) {
      int u = tid + p * 256;
      int r = u >> 4, c = (u & 15) * 2;
      v2h h2; h2[0] = (_Float16)aP[p].x; h2[1] = (_Float16)aP[p].y;
      *(v2h*)&Ash[r * ASTR + c] = h2;
    }
#pragma unroll
    for (int p = 0; p < NBP; ++p) {
      int u = tid + p * 256;
      int n, k;
      if (transB) { n = u >> 4; k = (u & 15) * 2; }
      else        { n = u % BN_; k = (u / BN_) * 2; }
      v2h h2; h2[0] = (_Float16)bP[p].x; h2[1] = (_Float16)bP[p].y;
      *(v2h*)&Bsh[n * BSTR + k] = h2;
    }
    __syncthreads();

    // Prefetch next K-tile while this tile computes
    if (k0 + BK < K) { loadA(k0 + BK); loadB(k0 + BK); }

    // A fragments: lanes 0-15 (rows mm) hold K 0..7 & 16..23; lanes 16-31 hold 8..15 & 24..31
    v16h af[2];
#pragma unroll
    for (int i = 0; i < 2; ++i) {
      const _Float16* ap = &Ash[(wr * 32 + i * 16 + mm) * ASTR];
      v8h lo  = *(const v8h*)(ap + hi * 8);
      v8h hi8 = *(const v8h*)(ap + 16 + hi * 8);
#pragma unroll
      for (int u = 0; u < 8; ++u) { af[i][u] = lo[u]; af[i][8 + u] = hi8[u]; }
    }
#pragma unroll
    for (int j = 0; j < NJ; ++j) {
      // B fragment: lanes 0-15 (cols mm) hold K 0..7 & 16..23; lanes 16-31 hold 8..15 & 24..31
      const _Float16* bp = &Bsh[(wc * (BN_ / 2) + j * 16 + mm) * BSTR + hi * 8];
      v8h lo  = *(const v8h*)(bp);
      v8h hi8 = *(const v8h*)(bp + 16);
      v16h bf;
#pragma unroll
      for (int u = 0; u < 8; ++u) { bf[u] = lo[u]; bf[8 + u] = hi8[u]; }
#pragma unroll
      for (int i = 0; i < 2; ++i)
        acc[i][j] = __builtin_amdgcn_wmma_f32_16x16x32_f16(
            false, af[i], false, bf, (short)0, acc[i][j], false, false);
    }
  }

  // C/D layout: lane -> col = mm; VGPR r -> row = r + 8*hi
#pragma unroll
  for (int i = 0; i < 2; ++i)
#pragma unroll
    for (int j = 0; j < NJ; ++j) {
      int col = tileN + wc * (BN_ / 2) + j * 16 + mm;
      if (col < N) {
#pragma unroll
        for (int r = 0; r < 8; ++r) {
          int row = tileM + wr * 32 + i * 16 + hi * 8 + r;
          float v = acc[i][j][r];
          if (bias) v += bias[col];
          if (Rb)   v += Rb[(size_t)row * ldc + col];
          if (relu) v = fmaxf(v, 0.f);
          Cb[(size_t)row * ldc + col] = v;
        }
      }
    }
}

// ---------------- Embedding + sinusoidal positional encoding ----------------
__global__ void __launch_bounds__(256)
embed_kernel(const int* __restrict__ idx, const float* __restrict__ tok,
             float* __restrict__ x)
{
  int i = blockIdx.x * 256 + threadIdx.x;      // over Mrows*Dm
  int row = i >> 9, d = i & (Dm - 1);
  int t = row & (Tm - 1);
  int token = idx[row];
  int de = d & ~1;
  float div = expf((float)de * (-9.210340371976184f / (float)Dm)); // -ln(10000)/D
  float ang = (float)t * div;
  float pe = (d & 1) ? cosf(ang) : sinf(ang);
  x[i] = tok[(size_t)token * Dm + d] + pe;
}

// ---------------- LayerNorm (one row per block, D=512) ----------------
__global__ void __launch_bounds__(256)
ln_kernel(const float* __restrict__ x, const float* __restrict__ g,
          const float* __restrict__ b, float* __restrict__ out)
{
  __shared__ float red[256];
  int row = blockIdx.x, tid = threadIdx.x;
  const float* xr = x + (size_t)row * Dm;
  float v0 = xr[tid], v1 = xr[tid + 256];
  red[tid] = v0 + v1;
  __syncthreads();
  for (int s = 128; s > 0; s >>= 1) { if (tid < s) red[tid] += red[tid + s]; __syncthreads(); }
  float mean = red[0] * (1.f / Dm);
  __syncthreads();
  float d0 = v0 - mean, d1 = v1 - mean;
  red[tid] = d0 * d0 + d1 * d1;
  __syncthreads();
  for (int s = 128; s > 0; s >>= 1) { if (tid < s) red[tid] += red[tid + s]; __syncthreads(); }
  float inv = rsqrtf(red[0] * (1.f / Dm) + 1e-5f);
  float* orow = out + (size_t)row * Dm;
  orow[tid]       = d0 * inv * g[tid] + b[tid];
  orow[tid + 256] = d1 * inv * g[tid + 256] + b[tid + 256];
}

// ---------------- Causal softmax over attention scores (scale 1/sqrt(DH)) ----------------
__global__ void __launch_bounds__(128)
attn_softmax(float* __restrict__ att)
{
  __shared__ float red[128];
  int r = blockIdx.x;                // over B*H*T rows
  int t = r & (Tm - 1);
  float* ar = att + (size_t)r * Tm;
  int c = threadIdx.x;
  float v = (c <= t) ? ar[c] * 0.125f : -1.0e30f;
  red[c] = v;
  __syncthreads();
  for (int s = 64; s > 0; s >>= 1) { if (c < s) red[c] = fmaxf(red[c], red[c + s]); __syncthreads(); }
  float mx = red[0];
  __syncthreads();
  float e = expf(v - mx);
  red[c] = e;
  __syncthreads();
  for (int s = 64; s > 0; s >>= 1) { if (c < s) red[c] += red[c + s]; __syncthreads(); }
  ar[c] = e / red[0];
}

// ---------------- Per-row NLL over V=200018 logits ----------------
__global__ void __launch_bounds__(256)
nll_kernel(const float* __restrict__ logits, const int* __restrict__ tgt,
           float* __restrict__ nll)
{
  __shared__ float red[256];
  int row = blockIdx.x, tid = threadIdx.x;
  const float* lr = logits + (size_t)row * Vm;
  float mx = -3.0e38f;
  for (int i = tid; i < Vm; i += 256) mx = fmaxf(mx, lr[i]);
  red[tid] = mx;
  __syncthreads();
  for (int s = 128; s > 0; s >>= 1) { if (tid < s) red[tid] = fmaxf(red[tid], red[tid + s]); __syncthreads(); }
  mx = red[0];
  __syncthreads();
  float sum = 0.f;
  for (int i = tid; i < Vm; i += 256) sum += expf(lr[i] - mx);
  red[tid] = sum;
  __syncthreads();
  for (int s = 128; s > 0; s >>= 1) { if (tid < s) red[tid] += red[tid + s]; __syncthreads(); }
  if (tid == 0) nll[row] = -(lr[tgt[row]] - mx - logf(red[0]));
}

__global__ void __launch_bounds__(256)
mean_kernel(const float* __restrict__ nll, float* __restrict__ out)
{
  __shared__ float red[256];
  int tid = threadIdx.x;
  float s = 0.f;
  for (int i = tid; i < Mrows; i += 256) s += nll[i];
  red[tid] = s;
  __syncthreads();
  for (int st = 128; st > 0; st >>= 1) { if (tid < st) red[tid] += red[tid + st]; __syncthreads(); }
  if (tid == 0) out[0] = red[0] * (1.f / Mrows);
}

// ---------------- Host orchestration ----------------
extern "C" void kernel_launch(void* const* d_in, const int* in_sizes, int n_in,
                              void* d_out, int out_size, void* d_ws, size_t ws_size,
                              hipStream_t stream)
{
  (void)in_sizes; (void)n_in; (void)out_size; (void)ws_size;

  const int*   idx     = (const int*)d_in[0];
  const int*   targets = (const int*)d_in[1];
  const float* tok_emb = (const float*)d_in[2];
  const float* wq    = (const float*)d_in[3];
  const float* wk    = (const float*)d_in[4];
  const float* wv    = (const float*)d_in[5];
  const float* wproj = (const float*)d_in[6];
  const float* bproj = (const float*)d_in[7];
  const float* ln1s  = (const float*)d_in[8];
  const float* ln1b  = (const float*)d_in[9];
  const float* ln2s  = (const float*)d_in[10];
  const float* ln2b  = (const float*)d_in[11];
  const float* w1    = (const float*)d_in[12];
  const float* b1    = (const float*)d_in[13];
  const float* w2    = (const float*)d_in[14];
  const float* b2    = (const float*)d_in[15];
  const float* lnfs  = (const float*)d_in[16];
  const float* lnfb  = (const float*)d_in[17];
  const float* outw  = (const float*)d_in[18];
  const float* outb  = (const float*)d_in[19];

  float* logits = (float*)d_out;
  float* loss   = logits + (size_t)Mrows * Vm;

  float* ws   = (float*)d_ws;
  float* x    = ws;                               // 1024*512
  float* h    = x    + (size_t)Mrows * Dm;
  float* q    = h    + (size_t)Mrows * Dm;
  float* kbuf = q    + (size_t)Mrows * Dm;
  float* vbuf = kbuf + (size_t)Mrows * Dm;
  float* obuf = vbuf + (size_t)Mrows * Dm;
  float* att  = obuf + (size_t)Mrows * Dm;        // B*H*T*T
  float* a1   = att  + (size_t)Bbat * Hm * Tm * Tm;  // 1024*2048
  float* nllb = a1   + (size_t)Mrows * 4 * Dm;    // 1024

  embed_kernel<<<Mrows * Dm / 256, 256, 0, stream>>>(idx, tok_emb, x);

  for (int l = 0; l < Lm; ++l) {
    ln_kernel<<<Mrows, 256, 0, stream>>>(x, ln1s + l * Dm, ln1b + l * Dm, h);

    // QKV projections: per-head [1024,512]x[512,64] -> columns h*64..h*64+63
    dim3 gq(Mrows / BM, 1, Hm);
    gemm_wmma<64><<<gq, 256, 0, stream>>>(h, Dm, 0, 0,
        wq + (size_t)l * Hm * Dm * DHm, DHm, 0, (long long)Dm * DHm, 0,
        q, Dm, 0, DHm, nullptr, nullptr, Mrows, DHm, Dm, Hm, 0);
    gemm_wmma<64><<<gq, 256, 0, stream>>>(h, Dm, 0, 0,
        wk + (size_t)l * Hm * Dm * DHm, DHm, 0, (long long)Dm * DHm, 0,
        kbuf, Dm, 0, DHm, nullptr, nullptr, Mrows, DHm, Dm, Hm, 0);
    gemm_wmma<64><<<gq, 256, 0, stream>>>(h, Dm, 0, 0,
        wv + (size_t)l * Hm * Dm * DHm, DHm, 0, (long long)Dm * DHm, 0,
        vbuf, Dm, 0, DHm, nullptr, nullptr, Mrows, DHm, Dm, Hm, 0);

    // scores[b,h] = Q[b,h] @ K[b,h]^T : [128,64]x[64,128]
    dim3 gs(1, 1, Bbat * Hm);
    gemm_wmma<128><<<gs, 256, 0, stream>>>(q, Dm, (long long)Tm * Dm, DHm,
        kbuf, Dm, (long long)Tm * Dm, DHm, 1,
        att, Tm, (long long)Hm * Tm * Tm, (long long)Tm * Tm,
        nullptr, nullptr, Tm, Tm, DHm, Hm, 0);

    attn_softmax<<<Bbat * Hm * Tm, 128, 0, stream>>>(att);

    // o[b,h] = attn[b,h] @ V[b,h] : [128,128]x[128,64]
    dim3 go(1, 1, Bbat * Hm);
    gemm_wmma<64><<<go, 256, 0, stream>>>(att, Tm, (long long)Hm * Tm * Tm, (long long)Tm * Tm,
        vbuf, Dm, (long long)Tm * Dm, DHm, 0,
        obuf, Dm, (long long)Tm * Dm, DHm,
        nullptr, nullptr, Tm, DHm, Tm, Hm, 0);

    // x = x + o @ wproj + bproj
    dim3 gp(Mrows / BM, Dm / 128, 1);
    gemm_wmma<128><<<gp, 256, 0, stream>>>(obuf, Dm, 0, 0,
        wproj + (size_t)l * Dm * Dm, Dm, 0, 0, 0,
        x, Dm, 0, 0,
        bproj + l * Dm, x, Mrows, Dm, Dm, 1, 0);

    ln_kernel<<<Mrows, 256, 0, stream>>>(x, ln2s + l * Dm, ln2b + l * Dm, h);

    // a1 = relu(h @ w1 + b1) : [1024,512]x[512,2048]
    dim3 g1(Mrows / BM, 4 * Dm / 128, 1);
    gemm_wmma<128><<<g1, 256, 0, stream>>>(h, Dm, 0, 0,
        w1 + (size_t)l * Dm * 4 * Dm, 4 * Dm, 0, 0, 0,
        a1, 4 * Dm, 0, 0,
        b1 + l * 4 * Dm, nullptr, Mrows, 4 * Dm, Dm, 1, 1);

    // x = x + a1 @ w2 + b2 : [1024,2048]x[2048,512]
    dim3 g2(Mrows / BM, Dm / 128, 1);
    gemm_wmma<128><<<g2, 256, 0, stream>>>(a1, 4 * Dm, 0, 0,
        w2 + (size_t)l * 4 * Dm * Dm, Dm, 0, 0, 0,
        x, Dm, 0, 0,
        b2 + l * Dm, x, Mrows, Dm, 4 * Dm, 1, 0);
  }

  ln_kernel<<<Mrows, 256, 0, stream>>>(x, lnfs, lnfb, h);

  // logits = h @ outw + outb : [1024,512]x[512,200018], ragged N
  dim3 gl(Mrows / BM, (Vm + 127) / 128, 1);
  gemm_wmma<128><<<gl, 256, 0, stream>>>(h, Dm, 0, 0,
      outw, Vm, 0, 0, 0,
      logits, Vm, 0, 0,
      outb, nullptr, Mrows, Vm, Dm, 1, 0);

  nll_kernel<<<Mrows, 256, 0, stream>>>(logits, targets, nllb);
  mean_kernel<<<1, 256, 0, stream>>>(nllb, loss);
}
